// MultiHeadSelfAttention_28887950033051
// MI455X (gfx1250) — compile-verified, matrix operand order checked by reading, not run on hardware
//
#include <hip/hip_runtime.h>

typedef __bf16 bf16;
typedef __bf16 v16bf __attribute__((ext_vector_type(16)));
typedef float  v8f   __attribute__((ext_vector_type(8)));

#define WMMA_BF16(a, b, c) \
  __builtin_amdgcn_wmma_f32_16x16x32_bf16(false, (a), false, (b), (short)0, (c), false, false)

// Load a 16x32 bf16 WMMA fragment in "NT" form: row r = lane&15 of `base`
// (row-major, leading dim `ld` elements), K contiguous starting at koff.
// Matches CDNA5 ISA 7.12.2 16-bit A layout (and B layout with row==N).
__device__ __forceinline__ v16bf load_nt(const bf16* base, int ld, int koff) {
  const int lane = threadIdx.x & 31;
  const int r    = lane & 15;
  const int kb   = (lane >> 4) << 3;
  const bf16* p  = base + (size_t)r * ld + koff + kb;
  v16bf f;
#pragma unroll
  for (int i = 0; i < 8; ++i) { f[i] = p[i]; f[8 + i] = p[16 + i]; }
  return f;
}

// Async-copy one 16B chunk per lane, global -> LDS (ASYNCcnt path).
// Tile layout: 64 rows x 32 bf16 cols, row-major in LDS; chunk c (0..255)
// covers row c>>2, bf16 cols (c&3)*8..+7; LDS byte offset = ldsbase + c*16.
// The generic pointer into s_lds is passed as an extra operand so the LDS
// object escapes into the asm and the "memory" clobber covers it (otherwise
// the compiler folds all loads from s_lds to undef — seen in round 2 disasm).
__device__ __forceinline__ void async_chunk(const bf16* gbase, int ldg, int k,
                                            int c, unsigned ldsbase, char* s_lds) {
  const bf16* g = gbase + (size_t)(c >> 2) * ldg + k + ((c & 3) << 3);
  unsigned lds  = ldsbase + (unsigned)c * 16u;
  char* dst     = s_lds + lds;   // escapes; %2 unused in template
  asm volatile("global_load_async_to_lds_b128 %0, %1, off"
               :: "v"(lds), "v"(g), "v"(dst) : "memory");
}

__device__ __forceinline__ void wait_async0() {
  asm volatile("s_wait_asynccnt 0x0" ::: "memory");
}

__global__ __launch_bounds__(256)
void cvt_f32_bf16(const float* __restrict__ in, bf16* __restrict__ out, int n) {
  int i = blockIdx.x * 256 + threadIdx.x;
  if (i < n) out[i] = (bf16)in[i];
}

// Shared blocked-GEMM mainloop: C(64x64) += A(64xK) * B(64xK)^T, K=1024.
// 4 waves; wave w owns rows w*16..+15 and all 64 cols (4 accumulators).
// K staged in 32-wide slabs via async LDS copies, double buffered.
__device__ __forceinline__ void gemm_mainloop(const bf16* agp, const bf16* bgp,
                                              char* s_lds, v8f acc[4]) {
  const int tid  = threadIdx.x;
  const int wave = tid >> 5;
  const int c0 = tid, c1 = tid + 128;

  // prologue: stage k=0 into buffer 0 (A at +0, B at +4096)
  async_chunk(agp, 1024, 0, c0, 0u, s_lds);
  async_chunk(agp, 1024, 0, c1, 0u, s_lds);
  async_chunk(bgp, 1024, 0, c0, 4096u, s_lds);
  async_chunk(bgp, 1024, 0, c1, 4096u, s_lds);

  unsigned buf = 0;
  for (int k = 0; k < 1024; k += 32) {
    wait_async0();
    __syncthreads();   // current buffer staged; prior iter's reads complete
    const unsigned cur = buf * 8192u;
    const int nk = k + 32;
    if (nk < 1024) {
      const unsigned nxt = 8192u - cur;
      async_chunk(agp, 1024, nk, c0, nxt, s_lds);
      async_chunk(agp, 1024, nk, c1, nxt, s_lds);
      async_chunk(bgp, 1024, nk, c0, nxt + 4096u, s_lds);
      async_chunk(bgp, 1024, nk, c1, nxt + 4096u, s_lds);
    }
    const bf16* sa = (const bf16*)(s_lds + cur) + wave * 16 * 32;
    const bf16* sb = (const bf16*)(s_lds + cur + 4096u);
    v16bf av = load_nt(sa, 32, 0);
#pragma unroll
    for (int nt = 0; nt < 4; ++nt) {
      v16bf bv = load_nt(sb + nt * 16 * 32, 32, 0);
      acc[nt] = WMMA_BF16(av, bv, acc[nt]);
    }
    buf ^= 1u;
  }
}

// out = x @ W^T (bf16 in/out, f32 accumulate). M=4096, N=1024, K=1024.
// grid (64, 16): 64x64 tile per block. N-block == one head (64 cols).
// mode 0/1: write [B,H,S,P]; mode 2: write V transposed [B,H,P,S].
__global__ __launch_bounds__(128)
void qkv_proj(const bf16* __restrict__ xbf, const bf16* __restrict__ wbf,
              bf16* __restrict__ outb, int mode) {
  __shared__ char s_lds[16384];
  const int lane  = threadIdx.x & 31;
  const int wave  = threadIdx.x >> 5;
  const int mBase = blockIdx.x << 6;
  const int h     = blockIdx.y;                 // nBase = h*64
  v8f acc[4] = {{0.f,0.f,0.f,0.f,0.f,0.f,0.f,0.f}, {0.f,0.f,0.f,0.f,0.f,0.f,0.f,0.f},
                {0.f,0.f,0.f,0.f,0.f,0.f,0.f,0.f}, {0.f,0.f,0.f,0.f,0.f,0.f,0.f,0.f}};
  gemm_mainloop(xbf + (size_t)mBase * 1024,
                wbf + ((size_t)h << 6) * 1024, s_lds, acc);

  const int col = lane & 15, hi = lane >> 4;
#pragma unroll
  for (int nt = 0; nt < 4; ++nt) {
    const int p = (nt << 4) + col;
#pragma unroll
    for (int r = 0; r < 8; ++r) {
      const int m    = mBase + (wave << 4) + (hi << 3) + r;
      const int bidx = m >> 11, s = m & 2047;   // S = 2048
      const int bh   = bidx * 16 + h;
      bf16 v = (bf16)acc[nt][r];
      if (mode == 2)
        outb[(((size_t)bh << 6 | p) << 11) + s] = v;   // [B,H,P,S]
      else
        outb[(((size_t)bh << 11 | s) << 6) + p] = v;   // [B,H,S,P]
    }
  }
}

// out = concat @ Wo^T + bo, fp32 output.
__global__ __launch_bounds__(128)
void out_proj(const bf16* __restrict__ cbf, const bf16* __restrict__ wobf,
              const float* __restrict__ bo, float* __restrict__ out) {
  __shared__ char s_lds[16384];
  const int lane  = threadIdx.x & 31;
  const int wave  = threadIdx.x >> 5;
  const int mBase = blockIdx.x << 6;
  const int nBase = blockIdx.y << 6;
  v8f acc[4] = {{0.f,0.f,0.f,0.f,0.f,0.f,0.f,0.f}, {0.f,0.f,0.f,0.f,0.f,0.f,0.f,0.f},
                {0.f,0.f,0.f,0.f,0.f,0.f,0.f,0.f}, {0.f,0.f,0.f,0.f,0.f,0.f,0.f,0.f}};
  gemm_mainloop(cbf + (size_t)mBase * 1024,
                wobf + (size_t)nBase * 1024, s_lds, acc);

  const int col = lane & 15, hi = lane >> 4;
#pragma unroll
  for (int nt = 0; nt < 4; ++nt) {
    const int n = nBase + (nt << 4) + col;
    const float bias = bo[n];
#pragma unroll
    for (int r = 0; r < 8; ++r) {
      const int m = mBase + (wave << 4) + (hi << 3) + r;
      out[(size_t)m * 1024 + n] = acc[nt][r] + bias;
    }
  }
}

#define SCHUNK 256

// Flash attention: one block (4 waves) = 16 query rows of one (b,h).
__global__ __launch_bounds__(128)
void attn_fa(const bf16* __restrict__ qb, const bf16* __restrict__ kbuf,
             const bf16* __restrict__ vtb, bf16* __restrict__ concat) {
  __shared__ float s_scores[16][SCHUNK + 4];
  __shared__ bf16  s_probs[16][SCHUNK];
  __shared__ float s_red[16][8];
  __shared__ float s_rowmax[16], s_rowscale[16], s_rowsum[16];

  const int tid  = threadIdx.x;
  const int lane = tid & 31;
  const int wave = tid >> 5;
  const int sblk = blockIdx.x & 127;
  const int h    = (blockIdx.x >> 7) & 15;
  const int b    = blockIdx.x >> 11;
  const int bh   = b * 16 + h;

  const bf16* qhead  = qb   + ((size_t)bh << 17);   // 2048*64 per head
  const bf16* khead  = kbuf + ((size_t)bh << 17);
  const bf16* vthead = vtb  + ((size_t)bh << 17);

  const bf16* qblock = qhead + ((size_t)(sblk << 4) << 6);
  const v16bf aq0 = load_nt(qblock, 64, 0);
  const v16bf aq1 = load_nt(qblock, 64, 32);

  if (tid < 16) { s_rowmax[tid] = -3.0e38f; s_rowsum[tid] = 0.f; s_rowscale[tid] = 1.f; }
  v8f oacc = {0.f, 0.f, 0.f, 0.f, 0.f, 0.f, 0.f, 0.f};
  __syncthreads();

  const int row = tid >> 3, sub = tid & 7;
  const int hi = lane >> 4, col = lane & 15;

  for (int tc = 0; tc < 2048; tc += SCHUNK) {
    for (int nt = wave; nt < 16; nt += 4) {
      v8f acc = {0.f, 0.f, 0.f, 0.f, 0.f, 0.f, 0.f, 0.f};
      const bf16* kt = khead + (size_t)(tc + (nt << 4)) * 64;
      v16bf b0 = load_nt(kt, 64, 0);
      acc = WMMA_BF16(aq0, b0, acc);
      v16bf b1 = load_nt(kt, 64, 32);
      acc = WMMA_BF16(aq1, b1, acc);
#pragma unroll
      for (int r = 0; r < 8; ++r)
        s_scores[(hi << 3) + r][(nt << 4) + col] = acc[r] * 0.125f; // 1/sqrt(64)
    }
    __syncthreads();
    float mx = -3.0e38f;
    for (int c = sub; c < SCHUNK; c += 8) mx = fmaxf(mx, s_scores[row][c]);
    s_red[row][sub] = mx;
    __syncthreads();
    if (tid < 16) {
      float cm = s_red[tid][0];
#pragma unroll
      for (int i = 1; i < 8; ++i) cm = fmaxf(cm, s_red[tid][i]);
      const float old = s_rowmax[tid];
      const float nm  = fmaxf(old, cm);
      const float sc  = __expf(old - nm);
      s_rowmax[tid] = nm; s_rowscale[tid] = sc; s_rowsum[tid] *= sc;
    }
    __syncthreads();
#pragma unroll
    for (int r = 0; r < 8; ++r) oacc[r] *= s_rowscale[(hi << 3) + r];
    {
      const float rm = s_rowmax[row];
      float ps = 0.f;
      for (int c = sub; c < SCHUNK; c += 8) {
        float e = __expf(s_scores[row][c] - rm);
        ps += e;
        s_probs[row][c] = (bf16)e;
      }
      s_red[row][sub] = ps;
    }
    __syncthreads();
    if (tid < 16) {
      float s = 0.f;
#pragma unroll
      for (int i = 0; i < 8; ++i) s += s_red[tid][i];
      s_rowsum[tid] += s;
    }
    const bf16* vrow = vthead + ((size_t)(wave << 4) << 11);
#pragma unroll
    for (int ks = 0; ks < SCHUNK; ks += 32) {
      v16bf pa = load_nt(&s_probs[0][0], SCHUNK, ks);
      v16bf vb = load_nt(vrow, 2048, tc + ks);
      oacc = WMMA_BF16(pa, vb, oacc);
    }
    __syncthreads();
  }

  const size_t outBase =
      ((size_t)b * 2048 + (sblk << 4)) * 1024 + h * 64 + (wave << 4) + col;
#pragma unroll
  for (int r = 0; r < 8; ++r) {
    const int m = (hi << 3) + r;
    concat[outBase + (size_t)m * 1024] = (bf16)(oacc[r] / s_rowsum[m]);
  }
}

extern "C" void kernel_launch(void* const* d_in, const int* in_sizes, int n_in,
                              void* d_out, int out_size, void* d_ws, size_t ws_size,
                              hipStream_t stream) {
  const float* x  = (const float*)d_in[0];
  const float* Wq = (const float*)d_in[1];
  const float* Wk = (const float*)d_in[2];
  const float* Wv = (const float*)d_in[3];
  const float* Wo = (const float*)d_in[4];
  const float* bo = (const float*)d_in[5];

  const int XN = 4096 * 1024;   // B*S*D
  const int WN = 1024 * 1024;   // D*D
  const int QN = 2 * 16 * 2048 * 64;

  bf16* xbf = (bf16*)d_ws;
  bf16* wqb = xbf + XN;
  bf16* wkb = wqb + WN;
  bf16* wvb = wkb + WN;
  bf16* wob = wvb + WN;
  bf16* qb  = wob + WN;
  bf16* kb  = qb + QN;
  bf16* vtb = kb + QN;
  bf16* cb  = vtb + QN;

  cvt_f32_bf16<<<XN / 256, 256, 0, stream>>>(x,  xbf, XN);
  cvt_f32_bf16<<<WN / 256, 256, 0, stream>>>(Wq, wqb, WN);
  cvt_f32_bf16<<<WN / 256, 256, 0, stream>>>(Wk, wkb, WN);
  cvt_f32_bf16<<<WN / 256, 256, 0, stream>>>(Wv, wvb, WN);
  cvt_f32_bf16<<<WN / 256, 256, 0, stream>>>(Wo, wob, WN);

  dim3 g(64, 16), blk(128);
  qkv_proj<<<g, blk, 0, stream>>>(xbf, wqb, qb, 0);
  qkv_proj<<<g, blk, 0, stream>>>(xbf, wkb, kb, 1);
  qkv_proj<<<g, blk, 0, stream>>>(xbf, wvb, vtb, 2);

  attn_fa<<<4096, 128, 0, stream>>>(qb, kb, vtb, cb);

  out_proj<<<g, blk, 0, stream>>>(cb, wob, bo, (float*)d_out);
}
